// LlamaAttention_25598005084771
// MI455X (gfx1250) — matrix operand order repaired and checked by reading, compiled, not run on hardware
//
#include <hip/hip_runtime.h>

// CDNA5 / gfx1250, wave32. bf16 WMMA path: v_wmma_f32_16x16x32_bf16,
// async LDS staging: global_load_async_to_lds_b128 + s_wait_asynccnt.
typedef __attribute__((ext_vector_type(16))) __bf16 v16bf;
typedef __attribute__((ext_vector_type(8)))  __bf16 v8bf;
typedef __attribute__((ext_vector_type(4)))  __bf16 v4bf;
typedef __attribute__((ext_vector_type(8)))  float  v8f;

#define NB   2
#define NQ   1024
#define NH   32
#define NKVH 8
#define NHD  64
#define BQ   (NB * NQ)

#define SHUF16(lo, hi) \
  __builtin_shufflevector(lo, hi, 0,1,2,3,4,5,6,7,8,9,10,11,12,13,14,15)

__device__ __forceinline__ __bf16 f2bf(float f) {
  unsigned u = __float_as_uint(f);
  u += 0x7fffu + ((u >> 16) & 1u);              // round-to-nearest-even
  unsigned short h = (unsigned short)(u >> 16);
  return __builtin_bit_cast(__bf16, h);
}

// ---------------------------------------------------------------------------
// Bulk fp32 -> bf16 (RNE), 4 elements/thread. Pays the conversion cost ONCE
// so the WMMA inner loops are pure load+wmma.
// ---------------------------------------------------------------------------
__global__ void __launch_bounds__(256) conv_bf16(
    const float4* __restrict__ s, v4bf* __restrict__ d, int n4)
{
  int i = blockIdx.x * 256 + threadIdx.x;
  if (i >= n4) return;
  float4 f = s[i];
  v4bf o = { f2bf(f.x), f2bf(f.y), f2bf(f.z), f2bf(f.w) };
  d[i] = o;
}

// ---------------------------------------------------------------------------
// C[M,N] = A[M,K] @ W[N,K]^T, bf16 inputs, fp32 WMMA accumulate.
// One wave computes a 16x64 strip (A-frag reused across 4 WMMAs / K-step).
// 8 waves per block -> 512 columns per block.
// A-frag (16x32): lane<16 rows m=lane, K {0..15}; lane>=16 K {16..31}.
// B-frag (32x16): lane = column, 16 contiguous K per half.
// ---------------------------------------------------------------------------
__global__ void __launch_bounds__(256) gemm_bf16(
    const __bf16* __restrict__ A, const __bf16* __restrict__ W,
    float* __restrict__ C, int M, int N, int K)
{
  const int wave    = threadIdx.x >> 5;
  const int lane    = threadIdx.x & 31;
  const int nblocks = N >> 9;                       // 512-col blocks
  const int mtile   = blockIdx.x / nblocks;
  const int nblk    = blockIdx.x - mtile * nblocks;
  const int row0    = mtile << 4;
  const int col0    = (nblk << 9) + (wave << 6);

  const int halfsel = lane >> 4;
  const int abase   = halfsel << 4;
  const int bbase   = halfsel << 4;
  const int l15     = lane & 15;

  const __bf16* __restrict__ arow = A + (size_t)(row0 + l15) * K + abase;
  const __bf16* __restrict__ wr0  = W + (size_t)(col0 + l15) * K + bbase;

  v8f acc[4] = {};
  for (int k0 = 0; k0 < K; k0 += 32) {
    __builtin_prefetch(arow + k0 + 128, 0, 1);      // global_prefetch_b8
    v8bf alo = *(const v8bf*)(arow + k0);
    v8bf ahi = *(const v8bf*)(arow + k0 + 8);
    v16bf af = SHUF16(alo, ahi);
#pragma unroll
    for (int c = 0; c < 4; ++c) {
      v16bf bfr = *(const v16bf*)(wr0 + (size_t)(c * 16) * K + k0);
      acc[c] = __builtin_amdgcn_wmma_f32_16x16x32_bf16(false, af, false, bfr,
                                                       (short)0, acc[c], false, false);
    }
  }

  const int rbase = row0 + (halfsel << 3);
#pragma unroll
  for (int c = 0; c < 4; ++c) {
    const int cn = col0 + c * 16 + l15;
#pragma unroll
    for (int r = 0; r < 8; ++r)
      C[(size_t)(rbase + r) * N + cn] = acc[c][r];
  }
}

// ---------------------------------------------------------------------------
// RoPE at positions pos_ids + LCK(=2), in place on fp32, plus bf16 copy.
// ---------------------------------------------------------------------------
__global__ void __launch_bounds__(256) rope_bf16(
    float* __restrict__ x, __bf16* __restrict__ xbf,
    const int* __restrict__ pos_ids, int nheads, int total)
{
  int idx = blockIdx.x * 256 + threadIdx.x;
  if (idx >= total) return;
  int i   = idx & 31;
  int h   = (idx >> 5) % nheads;
  int row = idx / (nheads << 5);                   // b*Q + q
  float pos  = (float)(pos_ids[row] + 2);
  float freq = pos * __powf(10000.0f, -(float)i / 32.0f);
  float s, c;
  __sincosf(freq, &s, &c);
  size_t off = (size_t)row * (nheads * 64) + (size_t)h * 64 + i;
  float x1 = x[off], x2 = x[off + 32];
  float o1 = x1 * c - x2 * s;
  float o2 = x2 * c + x1 * s;
  x[off] = o1;  x[off + 32] = o2;
  xbf[off] = f2bf(o1);  xbf[off + 32] = f2bf(o2);
}

// ---------------------------------------------------------------------------
// Cache entry 0: K0 -> bf16 [b,h,q,d]; V0 -> bf16 transposed [b,h,d,q]
// so PV B-frags are contiguous (both in global and after LDS staging).
// ---------------------------------------------------------------------------
__global__ void __launch_bounds__(256) conv_cache(
    const float* __restrict__ k0, const float* __restrict__ v0,
    __bf16* __restrict__ k0bf, __bf16* __restrict__ v0t)
{
  size_t idx = (size_t)blockIdx.x * 256 + threadIdx.x;   // exact: NB*NH*NQ*NHD
  k0bf[idx] = f2bf(k0[idx]);
  size_t d  = idx & 63;
  size_t q  = (idx >> 6) & 1023;
  size_t bh = idx >> 16;
  v0t[(bh * 64 + d) * 1024 + q] = f2bf(v0[idx]);
}

// ---------------------------------------------------------------------------
// Flash-style attention, one wave = one (b, h, 16-row query tile).
// K/V 32-key tiles double-buffered in LDS via async-to-LDS DMA (wave-private
// buffers -> no workgroup barriers; ASYNCcnt/DScnt waits only).
// ---------------------------------------------------------------------------
__global__ void __launch_bounds__(128) attn_fwd(
    const __bf16* __restrict__ qbf, const float* __restrict__ qf,
    const __bf16* __restrict__ k0bf, const __bf16* __restrict__ v0t,
    const float* __restrict__ k1, const float* __restrict__ v1,
    const float* __restrict__ kfp, const float* __restrict__ vfp,
    __bf16* __restrict__ outp)
{
  __shared__ alignas(32) __bf16 Kt[4][2][32 * 64];   // 32 keys x 64 d
  __shared__ alignas(32) __bf16 Vt[4][2][64 * 32];   // 64 d x 32 keys
  __shared__ alignas(32) float  Sb[4][16][32];
  __shared__ alignas(32) __bf16 Pb[4][16][32];
  __shared__ alignas(32) float  Ob[4][16][64];

  const int wave = threadIdx.x >> 5;
  const int lane = threadIdx.x & 31;
  const int tile = blockIdx.x * 4 + wave;
  const int q0   = (tile & 63) << 4;
  const int h    = (tile >> 6) & 31;
  const int b    = tile >> 11;
  const int kvh  = h >> 2;                          // N_REP = 4

  const int halfsel = lane >> 4;
  const int abase   = halfsel << 4;
  const int bbase   = halfsel << 4;
  const int l15     = lane & 15;
  const int rb      = halfsel << 3;
  const float scale = 0.125f;                       // 1/sqrt(64)

  const size_t bh = (size_t)b * NH + h;
  const __bf16* kb0 = k0bf + bh * ((size_t)NQ * NHD);
  const __bf16* vtb = v0t + bh * ((size_t)NHD * NQ);

  const unsigned kl_base0 = (unsigned)(size_t)&Kt[wave][0][0];
  const unsigned kl_base1 = (unsigned)(size_t)&Kt[wave][1][0];
  const unsigned vl_base0 = (unsigned)(size_t)&Vt[wave][0][0];
  const unsigned vl_base1 = (unsigned)(size_t)&Vt[wave][1][0];

  // async stage one 32-key K tile (4KB contiguous) + V tile (64 rows x 64B)
  auto stage = [&](int kb, int buf) {
    const unsigned klb = (buf ? kl_base1 : kl_base0) + lane * 16;
    const unsigned long long kga =
        (unsigned long long)(size_t)(kb0 + (size_t)kb * NHD) + lane * 16;
#pragma unroll
    for (int i = 0; i < 8; ++i) {
      unsigned la = klb + i * 512;
      unsigned long long ga = kga + i * 512;
      asm volatile("global_load_async_to_lds_b128 %0, %1, off"
                   :: "v"(la), "v"(ga) : "memory");
    }
    const int vrow = lane >> 2;
    const int vch  = (lane & 3) * 16;
    const unsigned vlb = (buf ? vl_base1 : vl_base0) + vch;
#pragma unroll
    for (int i = 0; i < 8; ++i) {
      int d = i * 8 + vrow;
      unsigned la = vlb + d * 64;
      unsigned long long ga =
          (unsigned long long)(size_t)(vtb + (size_t)d * NQ + kb) + vch;
      asm volatile("global_load_async_to_lds_b128 %0, %1, off"
                   :: "v"(la), "v"(ga) : "memory");
    }
  };

  // Q A-frags (d = 0..63 -> two K=32 fragments), vectorized 16B loads
  const __bf16* qrow = qbf + (size_t)(b * NQ + q0 + l15) * (NH * NHD) + h * NHD;
  v16bf qa0 = SHUF16(*(const v8bf*)(qrow + abase),
                     *(const v8bf*)(qrow + abase + 8));
  v16bf qa1 = SHUF16(*(const v8bf*)(qrow + 32 + abase),
                     *(const v8bf*)(qrow + 32 + abase + 8));

  float m_row = -1e30f, l_row = 0.0f;               // valid in lanes 0..15
  v8f oacc[4] = {};

  const int kend = q0 + 16;                         // causal limit
  stage(0, 0);

  for (int kb = 0; kb < kend; kb += 32) {
    const int  cur     = (kb >> 5) & 1;
    const bool hasnext = (kb + 32) < kend;

    // WAR: make sure last iteration's LDS reads of buf cur^1 retired
    asm volatile("s_wait_dscnt 0x0" ::: "memory");
    if (hasnext) {
      stage(kb + 32, cur ^ 1);
      asm volatile("s_wait_asynccnt 0x10" ::: "memory");   // cur tile arrived
    } else {
      asm volatile("s_wait_asynccnt 0x0" ::: "memory");
    }

    const __bf16* Kl = &Kt[wave][cur][0];
    const __bf16* Vl = &Vt[wave][cur][0];

    // ---- S = Q @ K0^T for 32 keys (two 16-col WMMA tiles, K=64 contraction)
#pragma unroll
    for (int t = 0; t < 2; ++t) {
      const __bf16* krow = Kl + (t * 16 + l15) * 64;
      v16bf kf0 = *(const v16bf*)(krow + bbase);
      v16bf kf1 = *(const v16bf*)(krow + 32 + bbase);
      v8f s = {};
      s = __builtin_amdgcn_wmma_f32_16x16x32_bf16(false, qa0, false, kf0, (short)0, s, false, false);
      s = __builtin_amdgcn_wmma_f32_16x16x32_bf16(false, qa1, false, kf1, (short)0, s, false, false);
      const int key = kb + t * 16 + l15;
      const int n = t * 16 + l15;
#pragma unroll
      for (int r = 0; r < 8; ++r) {
        const int qq = q0 + rb + r;
        Sb[wave][rb + r][n] = s[r] * scale + ((key <= qq) ? 0.0f : -1e9f);
      }
    }
    asm volatile("s_wait_dscnt 0x0" ::: "memory");

    // ---- online softmax update (one row per lane in 0..15)
    float cfac = 0.0f;
    if (lane < 16) {
      const int row = lane;
      float mx = m_row;
#pragma unroll 4
      for (int j = 0; j < 32; ++j) mx = fmaxf(mx, Sb[wave][row][j]);
      cfac = __expf(m_row - mx);
      float sum = 0.0f;
#pragma unroll 4
      for (int j = 0; j < 32; ++j) {
        float p = __expf(Sb[wave][row][j] - mx);
        sum += p;
        Pb[wave][row][j] = f2bf(p);
      }
      l_row = l_row * cfac + sum;
      m_row = mx;
    }
    asm volatile("s_wait_dscnt 0x0" ::: "memory");

    // ---- rescale accumulators; broadcast per-row factor via ds_bpermute
    float cf[8];
#pragma unroll
    for (int r = 0; r < 8; ++r)
      cf[r] = __int_as_float(
          __builtin_amdgcn_ds_bpermute((rb + r) << 2, __float_as_int(cfac)));
#pragma unroll
    for (int c = 0; c < 4; ++c)
#pragma unroll
      for (int r = 0; r < 8; ++r) oacc[c][r] *= cf[r];

    // ---- O += P @ V0
    const __bf16* prow = &Pb[wave][l15][0];
    v16bf pa = SHUF16(*(const v8bf*)(prow + abase),
                      *(const v8bf*)(prow + abase + 8));
#pragma unroll
    for (int c = 0; c < 4; ++c) {
      v16bf vb = *(const v16bf*)(Vl + (c * 16 + l15) * 32 + bbase);
      oacc[c] = __builtin_amdgcn_wmma_f32_16x16x32_bf16(false, pa, false, vb,
                                                        (short)0, oacc[c], false, false);
    }
  }

  // ---- stage accumulators to LDS for row-major finalize
#pragma unroll
  for (int c = 0; c < 4; ++c) {
    const int n = c * 16 + l15;
#pragma unroll
    for (int r = 0; r < 8; ++r) Ob[wave][rb + r][n] = oacc[c][r];
  }
  asm volatile("s_wait_dscnt 0x0" ::: "memory");

  // ---- tail entries (cache_k[1] diag + fresh roped k diag), normalize, emit bf16
  if (lane < 16) {
    const int row = l15;
    const int qq  = q0 + row;
    const float* qfr = qf  + (size_t)(b * NQ + qq) * (NH * NHD)  + h  * NHD;
    const float* k1r = k1  + (bh * NQ + qq) * NHD;
    const float* kfr = kfp + (size_t)(b * NQ + qq) * (NKVH * NHD) + kvh * NHD;
    float s0 = 0.0f, s1 = 0.0f;
#pragma unroll 8
    for (int d = 0; d < 64; ++d) {
      float qv = qfr[d];
      s0 += qv * k1r[d];
      s1 += qv * kfr[d];
    }
    s0 *= scale;  s1 *= scale;
    float mx  = fmaxf(m_row, fmaxf(s0, s1));
    float c0  = __expf(m_row - mx);
    float e0  = __expf(s0 - mx);
    float e1  = __expf(s1 - mx);
    float inv = 1.0f / (l_row * c0 + e0 + e1);
    const float* v1r = v1  + (bh * NQ + qq) * NHD;
    const float* vfr = vfp + (size_t)(b * NQ + qq) * (NKVH * NHD) + kvh * NHD;
    __bf16* orow = outp + (size_t)(b * NQ + qq) * (NH * NHD) + h * NHD;
#pragma unroll 8
    for (int d = 0; d < 64; ++d)
      orow[d] = f2bf((Ob[wave][row][d] * c0 + e0 * v1r[d] + e1 * vfr[d]) * inv);
  }
}

// ---------------------------------------------------------------------------
extern "C" void kernel_launch(void* const* d_in, const int* in_sizes, int n_in,
                              void* d_out, int out_size, void* d_ws, size_t ws_size,
                              hipStream_t stream)
{
  (void)in_sizes; (void)n_in; (void)out_size; (void)ws_size;

  const float* hs = (const float*)d_in[0];   // [B,Q,4096]
  const float* Wq = (const float*)d_in[1];   // [2048,4096]
  const float* Wk = (const float*)d_in[2];   // [512,4096]
  const float* Wv = (const float*)d_in[3];   // [512,4096]
  const float* Wo = (const float*)d_in[4];   // [2048,2048]  (C = A @ Wo^T)
  const float* ck = (const float*)d_in[5];   // [2,B,H,Q,HD]
  const float* cv = (const float*)d_in[6];   // [2,B,H,Q,HD]
  // d_in[7] = attention_mask -> computed in-kernel (causal 0/-1e9)
  const int*  pos = (const int*)d_in[8];     // [B,Q]
  float* outp = (float*)d_out;

  // workspace carve-out (~106 MB)
  char* p = (char*)d_ws;
  float*  qproj  = (float*)p;  p += (size_t)BQ * 2048 * 4;
  float*  kproj  = (float*)p;  p += (size_t)BQ * 512 * 4;
  float*  vproj  = (float*)p;  p += (size_t)BQ * 512 * 4;
  __bf16* qbf    = (__bf16*)p; p += (size_t)BQ * 2048 * 2;
  __bf16* kbf    = (__bf16*)p; p += (size_t)BQ * 512 * 2;
  __bf16* k0bf   = (__bf16*)p; p += (size_t)NB * NH * NQ * NHD * 2;
  __bf16* v0t    = (__bf16*)p; p += (size_t)NB * NH * NQ * NHD * 2;
  __bf16* attnbf = (__bf16*)p; p += (size_t)BQ * 2048 * 2;
  __bf16* hsbf   = (__bf16*)p; p += (size_t)BQ * 4096 * 2;
  __bf16* wqbf   = (__bf16*)p; p += (size_t)2048 * 4096 * 2;
  __bf16* wkbf   = (__bf16*)p; p += (size_t)512 * 4096 * 2;
  __bf16* wvbf   = (__bf16*)p; p += (size_t)512 * 4096 * 2;
  __bf16* wobf   = (__bf16*)p; p += (size_t)2048 * 2048 * 2;

  const size_t cache_ent = (size_t)NB * NH * NQ * NHD;   // one cache entry

  // one-time bf16 conversions (bandwidth pass; keeps WMMA loops pure)
  auto conv = [&](const float* s, __bf16* d, size_t n) {
    int n4 = (int)(n >> 2);
    conv_bf16<<<dim3((n4 + 255) / 256), dim3(256), 0, stream>>>(
        (const float4*)s, (v4bf*)d, n4);
  };
  conv(hs, hsbf, (size_t)BQ * 4096);
  conv(Wq, wqbf, (size_t)2048 * 4096);
  conv(Wk, wkbf, (size_t)512 * 4096);
  conv(Wv, wvbf, (size_t)512 * 4096);
  conv(Wo, wobf, (size_t)2048 * 2048);

  // QKV projections (bf16 WMMA, fp32 accumulate)
  gemm_bf16<<<dim3((BQ / 16) * (2048 / 512)), dim3(256), 0, stream>>>(hsbf, wqbf, qproj, BQ, 2048, 4096);
  gemm_bf16<<<dim3((BQ / 16) * (512 / 512)),  dim3(256), 0, stream>>>(hsbf, wkbf, kproj, BQ, 512, 4096);
  gemm_bf16<<<dim3((BQ / 16) * (512 / 512)),  dim3(256), 0, stream>>>(hsbf, wvbf, vproj, BQ, 512, 4096);

  // RoPE (in-place fp32 + bf16 copies)
  const int qtot = BQ * NH * 32;
  rope_bf16<<<dim3((qtot + 255) / 256), dim3(256), 0, stream>>>(qproj, qbf, pos, NH, qtot);
  const int ktot = BQ * NKVH * 32;
  rope_bf16<<<dim3((ktot + 255) / 256), dim3(256), 0, stream>>>(kproj, kbf, pos, NKVH, ktot);

  // cache entry-0 bf16 conversion (V transposed)
  conv_cache<<<dim3((unsigned)(cache_ent / 256)), dim3(256), 0, stream>>>(ck, cv, k0bf, v0t);

  // flash attention + tail merge (emits bf16 for O-proj)
  attn_fwd<<<dim3(NB * NH * (NQ / 16) / 4), dim3(128), 0, stream>>>(
      qbf, qproj, k0bf, v0t, ck + cache_ent, cv + cache_ent, kproj, vproj, attnbf);

  // output projection
  gemm_bf16<<<dim3((BQ / 16) * (2048 / 512)), dim3(256), 0, stream>>>(attnbf, wobf, outp, BQ, 2048, 2048);
}
